// SerriformBlock_10548439679755
// MI455X (gfx1250) — matrix operand order repaired and checked by reading, not compiled, not run on hardware
//
#include <hip/hip_runtime.h>
#include <hip/hip_bf16.h>

// ---------------------------------------------------------------------------
// SerriformBlock for MI455X (gfx1250, wave32, WMMA).
// Shapes: B=4, L=2048, D=1024, MD=256, K=5, T=B*L=8192.
// All heavy matmuls via v_wmma_f32_16x16x32_f16 (f16 in, f32 acc).
// GEMM tiling: block = 8 waves, each wave owns a 32(M) x 64(N) tile:
// two A fragments share four B fragments -> 8 WMMAs per 192B/lane loaded
// (~21 FLOP/B from cache). The recurrent scan is reduced to two 256x256
// GEMVs/step (4 persistent WGs); the D x MD recurrent output projection is
// hoisted into a parallel WMMA GEMM over all stored m2 states.
// ---------------------------------------------------------------------------

typedef __attribute__((ext_vector_type(16))) _Float16 v16h;
typedef __attribute__((ext_vector_type(8)))  float    v8f;

union Frag16 { v16h v; uint4 q[2]; };

__device__ __forceinline__ float act_apply(float v, int mode) {
  if (mode == 1) return v / (1.0f + __expf(-v));                      // silu
  if (mode == 2) return 0.5f * v * (1.0f + erff(v * 0.70710678118f)); // gelu (exact)
  return v;
}

// OUT[M,N] = act( A[M,K] @ W[N,K]^T + bias + add0 + add1 + add2 )
// A fragment (16x32 f16): lane<16 holds row (lane), K chunks {k0..k0+7},{k0+16..23}
//                         lane>=16 holds row (lane-16), chunks {k0+8..15},{k0+24..31}
// B fragment (32x16 f16): lane<16 holds col (lane), K chunks {k0..k0+7},{k0+16..23}
//                         lane>=16 holds col (lane-16), chunks {k0+8..15},{k0+24..31}
// C/D: VGPR r -> (M = r + 8*(lane>>4), N = lane&15)   [ISA 7.12.2]
__global__ void __launch_bounds__(256)
k_gemm_wmma(const _Float16* __restrict__ A, int lda,
            const _Float16* __restrict__ W, int ldw,
            const float* __restrict__ bias,
            const float* __restrict__ add0,
            const float* __restrict__ add1,
            const float* __restrict__ add2,
            float* __restrict__ outF, _Float16* __restrict__ outH,
            int N, int K, int mode)
{
  const int lane = threadIdx.x & 31;
  const int wave = threadIdx.x >> 5;
  const int hi   = lane >> 4;
  const int lo   = lane & 15;
  const int mBase = blockIdx.y * 256 + wave * 32;
  const int nBase = blockIdx.x * 64;

  v8f acc[2][4] = {};

  const _Float16* aRow0 = A + (size_t)(mBase + lo) * lda + hi * 8;
  const _Float16* aRow1 = aRow0 + (size_t)16 * lda;

  for (int k0 = 0; k0 < K; k0 += 32) {
    __builtin_prefetch((const void*)(aRow0 + k0 + 128), 0, 1); // global_prefetch_b8
    __builtin_prefetch((const void*)(aRow1 + k0 + 128), 0, 1);
    Frag16 a0, a1;
    a0.q[0] = *reinterpret_cast<const uint4*>(aRow0 + k0);
    a0.q[1] = *reinterpret_cast<const uint4*>(aRow0 + k0 + 16);
    a1.q[0] = *reinterpret_cast<const uint4*>(aRow1 + k0);
    a1.q[1] = *reinterpret_cast<const uint4*>(aRow1 + k0 + 16);
#pragma unroll
    for (int j = 0; j < 4; ++j) {
      const _Float16* wRow = W + (size_t)(nBase + j * 16 + lo) * ldw + hi * 8 + k0;
      Frag16 b;
      b.q[0] = *reinterpret_cast<const uint4*>(wRow);
      b.q[1] = *reinterpret_cast<const uint4*>(wRow + 16);
      acc[0][j] = __builtin_amdgcn_wmma_f32_16x16x32_f16(
          false, a0.v, false, b.v, (short)0, acc[0][j], false, false);
      acc[1][j] = __builtin_amdgcn_wmma_f32_16x16x32_f16(
          false, a1.v, false, b.v, (short)0, acc[1][j], false, false);
    }
  }

#pragma unroll
  for (int j = 0; j < 4; ++j) {
    const int n = nBase + j * 16 + lo;
    const float bv = bias ? bias[n] : 0.0f;
#pragma unroll
    for (int i = 0; i < 2; ++i) {
#pragma unroll
      for (int r = 0; r < 8; ++r) {
        const int m = mBase + i * 16 + r + hi * 8;
        const size_t idx = (size_t)m * N + n;
        float v = acc[i][j][r] + bv;
        if (add0) v += add0[idx];
        if (add1) v += add1[idx];
        if (add2) v += add2[idx];
        v = act_apply(v, mode);
        if (outF) outF[idx] = v;
        if (outH) outH[idx] = (_Float16)v;
      }
    }
  }
}

// one block per row of D elements
__global__ void __launch_bounds__(256)
k_rmsnorm(const float* __restrict__ x, const float* __restrict__ w,
          float* __restrict__ outF, _Float16* __restrict__ outH, int D)
{
  __shared__ float red[8];
  __shared__ float scale_s;
  const size_t base = (size_t)blockIdx.x * D;
  float ss = 0.f;
  for (int d = threadIdx.x; d < D; d += 256) { float v = x[base + d]; ss += v * v; }
  for (int off = 16; off > 0; off >>= 1) ss += __shfl_xor(ss, off, 32);
  if ((threadIdx.x & 31) == 0) red[threadIdx.x >> 5] = ss;
  __syncthreads();
  if (threadIdx.x == 0) {
    float s = 0.f;
    for (int i = 0; i < 8; ++i) s += red[i];
    scale_s = rsqrtf(s / (float)D + 1e-6f);
  }
  __syncthreads();
  const float sc = scale_s;
  for (int d = threadIdx.x; d < D; d += 256) {
    float o = w[d] * x[base + d] * sc;
    outF[base + d] = o;
    if (outH) outH[base + d] = (_Float16)o;
  }
}

// causal depthwise conv (K=5, dil=1) over xn[b,l,d]; output f16 [t,d]
__global__ void __launch_bounds__(256)
k_dwconv(const float* __restrict__ xn, const float* __restrict__ kern,
         const float* __restrict__ kbias, _Float16* __restrict__ outH,
         int L, int D)
{
  const size_t idx = (size_t)blockIdx.x * 256 + threadIdx.x;
  const int d = (int)(idx % D);
  const size_t t = idx / D;
  const int l = (int)(t % L);
  const size_t rowBase = (t - (size_t)l) * D + d; // start of batch b, channel d
  float s = kbias[d];
#pragma unroll
  for (int k = 0; k < 5; ++k) {
    const int ll = l - 4 + k;
    if (ll >= 0) s += xn[rowBase + (size_t)ll * D] * kern[d * 5 + k];
  }
  outH[idx] = (_Float16)s;
}

// persistent scan: one block per batch, 512 threads.
// threads [0,256) compute f[j], [256,512) compute u[j]; m kept in LDS.
__global__ void __launch_bounds__(512)
k_scan(const float* __restrict__ fx, const float* __restrict__ ux,
       const float* __restrict__ xm,
       const _Float16* __restrict__ Wfm, const _Float16* __restrict__ Wum, int wld,
       const float* __restrict__ mem0,
       _Float16* __restrict__ m2h, float* __restrict__ memOut, int L)
{
  __shared__ float m_s[256];
  __shared__ float f_s[256];
  __shared__ float u_s[256];
  const int b   = blockIdx.x;
  const int tid = threadIdx.x;
  const int j   = tid & 255;
  const int sel = tid >> 8;
  if (tid < 256) m_s[tid] = mem0[b * 256 + tid];
  __syncthreads();
  const _Float16* wrow = (sel ? Wum : Wfm) + (size_t)j * wld;
  const float*    pre  = (sel ? ux : fx);
  float* gdst = sel ? u_s : f_s;
  for (int l = 0; l < L; ++l) {
    const size_t t = (size_t)b * L + l;
    float acc = pre[t * 256 + j];
#pragma unroll 8
    for (int k = 0; k < 256; ++k) acc += m_s[k] * (float)wrow[k];
    gdst[j] = 1.0f / (1.0f + __expf(-acc));
    __syncthreads();
    if (tid < 256) {
      const float m2 = f_s[tid] * m_s[tid] + u_s[tid] * xm[t * 256 + tid];
      m_s[tid] = m2;
      m2h[t * 256 + tid] = (_Float16)m2;
    }
    __syncthreads();
  }
  if (tid < 256) memOut[b * 256 + tid] = m_s[tid];
}

// router: softmax(y @ router_w.T + router_b); one wave per token
__global__ void __launch_bounds__(256)
k_router(const float* __restrict__ y, const float* __restrict__ rw,
         const float* __restrict__ rb, float* __restrict__ r, int D)
{
  const int t = blockIdx.x * 8 + (threadIdx.x >> 5);
  const int lane = threadIdx.x & 31;
  const float* row = y + (size_t)t * D;
  float a0 = 0.f, a1 = 0.f;
  for (int c = lane; c < D; c += 32) {
    const float v = row[c];
    a0 += v * rw[c];
    a1 += v * rw[D + c];
  }
  for (int off = 16; off > 0; off >>= 1) {
    a0 += __shfl_xor(a0, off, 32);
    a1 += __shfl_xor(a1, off, 32);
  }
  if (lane == 0) {
    a0 += rb[0]; a1 += rb[1];
    const float mx = fmaxf(a0, a1);
    const float e0 = __expf(a0 - mx), e1 = __expf(a1 - mx);
    const float s = e0 + e1;
    r[t * 2]     = e0 / s;
    r[t * 2 + 1] = e1 / s;
  }
}

// y += r0*e0 + r1*e1  (in place)
__global__ void __launch_bounds__(256)
k_combine(float* __restrict__ y, const float* __restrict__ e0,
          const float* __restrict__ e1, const float* __restrict__ r, int D)
{
  const size_t idx = (size_t)blockIdx.x * 256 + threadIdx.x;
  const size_t t = idx / D;
  y[idx] = y[idx] + r[t * 2] * e0[idx] + r[t * 2 + 1] * e1[idx];
}

__global__ void __launch_bounds__(256)
k_cvt(const float* __restrict__ src, _Float16* __restrict__ dst, size_t n)
{
  const size_t idx = (size_t)blockIdx.x * 256 + threadIdx.x;
  if (idx < n) dst[idx] = (_Float16)src[idx];
}

extern "C" void kernel_launch(void* const* d_in, const int* in_sizes, int n_in,
                              void* d_out, int out_size, void* d_ws, size_t ws_size,
                              hipStream_t stream) {
  (void)in_sizes; (void)n_in; (void)out_size; (void)ws_size;
  const int Bn = 4, L = 2048, D = 1024, MD = 256;
  const int T = Bn * L;                       // 8192 tokens
  const int DM = D + MD;                      // 1280

  const float* x      = (const float*)d_in[0];
  const float* mem0   = (const float*)d_in[1];
  const float* norm_w = (const float*)d_in[2];
  const float* dw_k   = (const float*)d_in[3];
  const float* dw_b   = (const float*)d_in[4];
  const float* pw_w   = (const float*)d_in[5];
  const float* pw_b   = (const float*)d_in[6];
  const float* Win_w  = (const float*)d_in[7];
  const float* Win_b  = (const float*)d_in[8];
  const float* Wf_w   = (const float*)d_in[9];
  const float* Wf_b   = (const float*)d_in[10];
  const float* Wu_w   = (const float*)d_in[11];
  const float* Wu_b   = (const float*)d_in[12];
  const float* Wo_w   = (const float*)d_in[13];
  const float* Wo_b   = (const float*)d_in[14];
  const float* rt_w   = (const float*)d_in[15];
  const float* rt_b   = (const float*)d_in[16];
  const float* e0w1   = (const float*)d_in[17];
  const float* e0b1   = (const float*)d_in[18];
  const float* e0w2   = (const float*)d_in[19];
  const float* e0b2   = (const float*)d_in[20];
  const float* e1w1   = (const float*)d_in[21];
  const float* e1b1   = (const float*)d_in[22];
  const float* e1w2   = (const float*)d_in[23];
  const float* e1b2   = (const float*)d_in[24];
  const float* fn_w   = (const float*)d_in[25];
  const float* dn_w   = (const float*)d_in[26];
  const float* dn_b   = (const float*)d_in[27];
  const float* up_w   = (const float*)d_in[28];
  const float* up_b   = (const float*)d_in[29];

  float* out_f = (float*)d_out;
  float* memOut = out_f + (size_t)T * D;

  // --- bump allocator over d_ws ---
  size_t off = 0;
  auto alloc = [&](size_t bytes) -> void* {
    void* p = (char*)d_ws + off;
    off += (bytes + 255) & ~(size_t)255;
    return p;
  };
  float*     xn     = (float*)    alloc((size_t)T * D * 4);   // reused as y2
  _Float16*  xn_h   = (_Float16*) alloc((size_t)T * D * 2);   // reused as y2_h
  _Float16*  conv_h = (_Float16*) alloc((size_t)T * D * 2);
  float*     fx     = (float*)    alloc((size_t)T * MD * 4);
  float*     ux     = (float*)    alloc((size_t)T * MD * 4);
  float*     xm     = (float*)    alloc((size_t)T * MD * 4);
  float*     ox     = (float*)    alloc((size_t)T * D * 4);   // reused as e1
  float*     xconv  = (float*)    alloc((size_t)T * D * 4);   // reused as e0
  _Float16*  m2h    = (_Float16*) alloc((size_t)T * MD * 2);
  float*     y      = (float*)    alloc((size_t)T * D * 4);
  _Float16*  y_h    = (_Float16*) alloc((size_t)T * D * 2);
  float*     rprob  = (float*)    alloc((size_t)T * 2 * 4);
  _Float16*  hid_h  = (_Float16*) alloc((size_t)T * D * 2);   // expert hidden (reused)
  _Float16*  hd_h   = (_Float16*) alloc((size_t)T * MD * 2);  // ffn hidden
  _Float16*  Wf_h   = (_Float16*) alloc((size_t)MD * DM * 2);
  _Float16*  Wu_h   = (_Float16*) alloc((size_t)MD * DM * 2);
  _Float16*  Wo_h   = (_Float16*) alloc((size_t)D * DM * 2);
  _Float16*  Win_h  = (_Float16*) alloc((size_t)MD * D * 2);
  _Float16*  pw_h   = (_Float16*) alloc((size_t)D * D * 2);
  _Float16*  e0w1h  = (_Float16*) alloc((size_t)D * D * 2);
  _Float16*  e0w2h  = (_Float16*) alloc((size_t)D * D * 2);
  _Float16*  e1w1h  = (_Float16*) alloc((size_t)D * D * 2);
  _Float16*  e1w2h  = (_Float16*) alloc((size_t)D * D * 2);
  _Float16*  dn_h   = (_Float16*) alloc((size_t)MD * D * 2);
  _Float16*  up_h   = (_Float16*) alloc((size_t)D * MD * 2);
  float*     y2     = xn;      // alias (xn dead after y is formed)
  _Float16*  y2_h   = xn_h;    // alias
  float*     e0buf  = xconv;   // alias
  float*     e1buf  = ox;      // alias

  auto cvt = [&](const float* s, _Float16* dst, size_t n) {
    k_cvt<<<dim3((unsigned)((n + 255) / 256)), dim3(256), 0, stream>>>(s, dst, n);
  };
  auto gemm = [&](const _Float16* A, int lda, const _Float16* W, int ldw,
                  const float* bias, const float* a0, const float* a1, const float* a2,
                  float* oF, _Float16* oH, int N, int K, int mode) {
    dim3 g((unsigned)(N / 64), (unsigned)(T / 256));
    k_gemm_wmma<<<g, dim3(256), 0, stream>>>(A, lda, W, ldw, bias, a0, a1, a2,
                                             oF, oH, N, K, mode);
  };

  // ---- weight conversions (f32 -> f16) ----
  cvt(Wf_w,  Wf_h,  (size_t)MD * DM);
  cvt(Wu_w,  Wu_h,  (size_t)MD * DM);
  cvt(Wo_w,  Wo_h,  (size_t)D * DM);
  cvt(Win_w, Win_h, (size_t)MD * D);
  cvt(pw_w,  pw_h,  (size_t)D * D);
  cvt(e0w1,  e0w1h, (size_t)D * D);
  cvt(e0w2,  e0w2h, (size_t)D * D);
  cvt(e1w1,  e1w1h, (size_t)D * D);
  cvt(e1w2,  e1w2h, (size_t)D * D);
  cvt(dn_w,  dn_h,  (size_t)MD * D);
  cvt(up_w,  up_h,  (size_t)D * MD);

  // ---- 1. RMSNorm ----
  k_rmsnorm<<<dim3(T), dim3(256), 0, stream>>>(x, norm_w, xn, xn_h, D);

  // ---- 2. depthwise causal conv -> f16 ----
  k_dwconv<<<dim3((unsigned)((size_t)T * D / 256)), dim3(256), 0, stream>>>(
      xn, dw_k, dw_b, conv_h, L, D);

  // ---- 3. parallel projections (WMMA) ----
  gemm(xn_h, D, Wf_h,  DM, Wf_b,  nullptr, nullptr, nullptr, fx, nullptr, MD, D, 0);
  gemm(xn_h, D, Wu_h,  DM, Wu_b,  nullptr, nullptr, nullptr, ux, nullptr, MD, D, 0);
  gemm(xn_h, D, Wo_h,  DM, Wo_b,  nullptr, nullptr, nullptr, ox, nullptr, D,  D, 0);
  gemm(xn_h, D, Win_h, D,  Win_b, nullptr, nullptr, nullptr, xm, nullptr, MD, D, 1); // silu
  gemm(conv_h, D, pw_h, D, pw_b,  nullptr, nullptr, nullptr, xconv, nullptr, D, D, 0);

  // ---- 4. recurrent scan (serial core only) ----
  k_scan<<<dim3(Bn), dim3(512), 0, stream>>>(fx, ux, xm,
                                             Wf_h + D, Wu_h + D, DM,
                                             mem0, m2h, memOut, L);

  // ---- 5. hoisted recurrent output projection, fused y = m2@Wo_m^T + xconv + xn + ox
  gemm(m2h, MD, Wo_h + D, DM, nullptr, xconv, xn, ox, y, y_h, D, MD, 0);

  // ---- 6. router ----
  k_router<<<dim3(T / 8), dim3(256), 0, stream>>>(y, rt_w, rt_b, rprob, D);

  // ---- 7. experts (WMMA), combine, final norm ----
  gemm(y_h, D, e0w1h, D, e0b1, nullptr, nullptr, nullptr, nullptr, hid_h, D, D, 1);
  gemm(hid_h, D, e0w2h, D, e0b2, nullptr, nullptr, nullptr, e0buf, nullptr, D, D, 0);
  gemm(y_h, D, e1w1h, D, e1b1, nullptr, nullptr, nullptr, nullptr, hid_h, D, D, 1);
  gemm(hid_h, D, e1w2h, D, e1b2, nullptr, nullptr, nullptr, e1buf, nullptr, D, D, 0);
  k_combine<<<dim3((unsigned)((size_t)T * D / 256)), dim3(256), 0, stream>>>(
      y, e0buf, e1buf, rprob, D);
  k_rmsnorm<<<dim3(T), dim3(256), 0, stream>>>(y, fn_w, y2, y2_h, D);

  // ---- 8. FFN: gelu(y2@down^T+b) @ up^T + up_b + y2 + resid -> d_out ----
  gemm(y2_h, D, dn_h, D, dn_b, nullptr, nullptr, nullptr, nullptr, hd_h, MD, D, 2); // gelu
  gemm(hd_h, MD, up_h, MD, up_b, y2, x, nullptr, out_f, nullptr, D, MD, 0);
}
